// GNN_58282706206726
// MI455X (gfx1250) — matrix operand order repaired and checked by reading, not compiled, not run on hardware
//
#include <hip/hip_runtime.h>
#include <hip/hip_bf16.h>
#include <math.h>

typedef __attribute__((ext_vector_type(16))) __bf16 v16bf;
typedef __attribute__((ext_vector_type(8)))  float  v8f;

#define NNODES 50000
#define NEDGES 1600000
#define HID    128

// ---------------------------------------------------------------------------
// helpers
// ---------------------------------------------------------------------------
__device__ __forceinline__ void split16(const float (&f)[16], v16bf& hi, v16bf& lo) {
#pragma unroll
    for (int i = 0; i < 16; i++) {
        __bf16 h = (__bf16)f[i];
        hi[i] = h;
        lo[i] = (__bf16)(f[i] - (float)h);
    }
}

// A-fragment pattern: 8 floats at row[base..base+7], 8 at row[base+16..base+23]
__device__ __forceinline__ void load16_a(const float* __restrict__ row, int base, float (&f)[16]) {
    const float4* p0 = (const float4*)(row + base);
    const float4* p1 = (const float4*)(row + base + 16);
    float4 v0 = p0[0], v1 = p0[1], v2 = p1[0], v3 = p1[1];
    f[0]=v0.x; f[1]=v0.y; f[2]=v0.z; f[3]=v0.w;
    f[4]=v1.x; f[5]=v1.y; f[6]=v1.z; f[7]=v1.w;
    f[8]=v2.x; f[9]=v2.y; f[10]=v2.z; f[11]=v2.w;
    f[12]=v3.x; f[13]=v3.y; f[14]=v3.z; f[15]=v3.w;
}

// B-fragment pattern: 16 contiguous floats at row[base..base+15]
__device__ __forceinline__ void load16_b(const float* __restrict__ row, int base, float (&f)[16]) {
    const float4* p = (const float4*)(row + base);
    float4 v0 = p[0], v1 = p[1], v2 = p[2], v3 = p[3];
    f[0]=v0.x; f[1]=v0.y; f[2]=v0.z; f[3]=v0.w;
    f[4]=v1.x; f[5]=v1.y; f[6]=v1.z; f[7]=v1.w;
    f[8]=v2.x; f[9]=v2.y; f[10]=v2.z; f[11]=v2.w;
    f[12]=v3.x; f[13]=v3.y; f[14]=v3.z; f[15]=v3.w;
}

// ---------------------------------------------------------------------------
// WMMA GEMM:  Y[M x (nTiles*16)] = (X1 (+ X2)) @ W^T + bias  [, GELU+LayerNorm]
// K fixed at 128. Block = nTiles waves; each wave = one 16-col tile of a
// 16-row block. bf16x3 split emulation of fp32 matmul, fp32 accumulate.
// ---------------------------------------------------------------------------
__global__ void k_gemm_wmma(const float* __restrict__ X1, const float* __restrict__ X2,
                            const float* __restrict__ W,  const float* __restrict__ bias,
                            const float* __restrict__ lnw, const float* __restrict__ lnb,
                            float* __restrict__ Y, int nTiles, int fuseGeluLN)
{
    __shared__ float sm[16][HID + 4];
    __shared__ float smu[16], srs[16];

    const int tid  = threadIdx.x;
    const int wave = tid >> 5;          // which N tile
    const int lane = tid & 31;
    const int half = lane >> 4;         // 0: lanes 0-15, 1: lanes 16-31
    const int r    = lane & 15;
    const int tileM = blockIdx.x;
    const int col  = wave * 16 + r;     // output column (== W row)

    const float* xrow  = X1 + (size_t)(tileM * 16 + r) * HID;
    const float* x2row = X2 ? (X2 + (size_t)(tileM * 16 + r) * HID) : nullptr;
    const float* wrow  = W  + (size_t)col * HID;

    v8f acc = {};
#pragma unroll
    for (int kc = 0; kc < 4; kc++) {
        float af[16], wf[16];
        load16_a(xrow, kc * 32 + half * 8, af);
        if (x2row) {
            float a2[16];
            load16_a(x2row, kc * 32 + half * 8, a2);
#pragma unroll
            for (int i = 0; i < 16; i++) af[i] += a2[i];
        }
        load16_a(wrow, kc * 32 + half * 8, wf);

        v16bf ahi, alo, whi, wlo;
        split16(af, ahi, alo);
        split16(wf, whi, wlo);

        acc = __builtin_amdgcn_wmma_f32_16x16x32_bf16(false, ahi, false, whi, (short)0, acc, false, false);
        acc = __builtin_amdgcn_wmma_f32_16x16x32_bf16(false, ahi, false, wlo, (short)0, acc, false, false);
        acc = __builtin_amdgcn_wmma_f32_16x16x32_bf16(false, alo, false, whi, (short)0, acc, false, false);
    }

    const int ldY = nTiles * 16;
    const float b = bias[col];

    if (!fuseGeluLN) {
#pragma unroll
        for (int v = 0; v < 8; v++) {
            int vrow = v + half * 8;
            Y[(size_t)(tileM * 16 + vrow) * ldY + col] = acc[v] + b;
        }
    } else {
        float g[8];
#pragma unroll
        for (int v = 0; v < 8; v++) {
            int vrow = v + half * 8;
            float y  = acc[v] + b;
            float ge = 0.5f * y * (1.0f + erff(y * 0.70710678118654752f));  // exact GELU
            g[v] = ge;
            sm[vrow][col] = ge;
        }
        __syncthreads();
        if (tid < 16) {
            float s = 0.f, q = 0.f;
#pragma unroll 4
            for (int c = 0; c < HID; c++) { float t = sm[tid][c]; s += t; q += t * t; }
            float mu  = s * (1.0f / HID);
            float var = q * (1.0f / HID) - mu * mu;
            smu[tid] = mu;
            srs[tid] = rsqrtf(var + 1e-5f);
        }
        __syncthreads();
        const float lw = lnw[col], lb = lnb[col];
#pragma unroll
        for (int v = 0; v < 8; v++) {
            int vrow = v + half * 8;
            Y[(size_t)(tileM * 16 + vrow) * ldY + col] =
                (g[v] - smu[vrow]) * srs[vrow] * lw + lb;
        }
    }
}

// ---------------------------------------------------------------------------
// per-layer scratch init: aggr[N*H] = 0, segmax/segsum[N] = 0
// ---------------------------------------------------------------------------
__global__ void k_init(float* __restrict__ aggr, float* __restrict__ segmax,
                       float* __restrict__ segsum)
{
    int i = blockIdx.x * blockDim.x + threadIdx.x;
    if (i < NNODES * HID) aggr[i] = 0.f;
    if (i < NNODES) { segmax[i] = 0.f; segsum[i] = 0.f; }
}

// segment max of positive floats via order-preserving uint atomicMax
__global__ void k_segmax(const int* __restrict__ dst, const float* __restrict__ a,
                         unsigned* __restrict__ segmax_bits)
{
    int e = blockIdx.x * blockDim.x + threadIdx.x;
    if (e < NEDGES)
        atomicMax(&segmax_bits[dst[e]], __float_as_uint(a[e]));
}

// segment sum of e = a / max_a  (== exp(log a - max(log a)))
__global__ void k_segsum(const int* __restrict__ dst, const float* __restrict__ a,
                         const float* __restrict__ segmax, float* __restrict__ segsum)
{
    int e = blockIdx.x * blockDim.x + threadIdx.x;
    if (e < NEDGES)
        atomicAdd(&segsum[dst[e]], a[e] / segmax[dst[e]]);
}

// one wave per edge: lane moves float4 => 32 lanes * 4 = 128 features
__global__ void k_aggregate(const int* __restrict__ src, const int* __restrict__ dst,
                            const float* __restrict__ a,
                            const float* __restrict__ segmax, const float* __restrict__ segsum,
                            const float* __restrict__ rep, float* __restrict__ aggr)
{
    long long gid = (long long)blockIdx.x * blockDim.x + threadIdx.x;
    int e    = (int)(gid >> 5);
    int lane = (int)(gid & 31);
    if (e >= NEDGES) return;

    int s = src[e], d = dst[e];
    float att = (a[e] / segmax[d]) / (segsum[d] + 1e-16f);

    float4 v = ((const float4*)(rep + (size_t)s * HID))[lane];
    float* o = aggr + (size_t)d * HID + lane * 4;
    atomicAdd(o + 0, v.x * att);
    atomicAdd(o + 1, v.y * att);
    atomicAdd(o + 2, v.z * att);
    atomicAdd(o + 3, v.w * att);
}

// ---------------------------------------------------------------------------
extern "C" void kernel_launch(void* const* d_in, const int* in_sizes, int n_in,
                              void* d_out, int out_size, void* d_ws, size_t ws_size,
                              hipStream_t stream)
{
    const float* node_attr = (const float*)d_in[0];
    const int*   eidx      = (const int*)d_in[1];     // [2, E]
    const float* adv       = (const float*)d_in[3];   // [2, E]
    const float* W_inp     = (const float*)d_in[4];
    const float* b_inp     = (const float*)d_in[5];
    const float* W_a       = (const float*)d_in[6];   // [2,128,128]
    const float* b_a       = (const float*)d_in[7];   // [2,128]
    const float* ln_w      = (const float*)d_in[8];
    const float* ln_b      = (const float*)d_in[9];
    const float* W_out     = (const float*)d_in[10];  // [64,128]
    const float* b_out     = (const float*)d_in[11];
    float*       out       = (float*)d_out;

    const int* esrc = eidx;
    const int* edst = eidx + NEDGES;

    float* repA   = (float*)d_ws;
    float* repB   = repA + (size_t)NNODES * HID;
    float* aggr   = repB + (size_t)NNODES * HID;
    float* segmax = aggr + (size_t)NNODES * HID;
    float* segsum = segmax + NNODES;

    const int mBlocks = NNODES / 16;  // 3125, exact

    // node_rep = node_attr @ W_inp^T + b_inp
    k_gemm_wmma<<<mBlocks, 256, 0, stream>>>(node_attr, nullptr, W_inp, b_inp,
                                             nullptr, nullptr, repA, 8, 0);

    float* cur = repA;
    float* nxt = repB;
    for (int l = 0; l < 2; l++) {
        const float* al = adv + (size_t)l * NEDGES;

        k_init<<<(NNODES * HID + 255) / 256, 256, 0, stream>>>(aggr, segmax, segsum);
        k_segmax<<<(NEDGES + 255) / 256, 256, 0, stream>>>(edst, al, (unsigned*)segmax);
        k_segsum<<<(NEDGES + 255) / 256, 256, 0, stream>>>(edst, al, segmax, segsum);
        k_aggregate<<<(int)(((long long)NEDGES * 32 + 255) / 256), 256, 0, stream>>>(
            esrc, edst, al, segmax, segsum, cur, aggr);

        // node_rep = LN(gelu((aggr + node_rep) @ W_a[l]^T + b_a[l]))
        k_gemm_wmma<<<mBlocks, 256, 0, stream>>>(aggr, cur,
                                                 W_a + (size_t)l * HID * HID, b_a + (size_t)l * HID,
                                                 ln_w + (size_t)l * HID, ln_b + (size_t)l * HID,
                                                 nxt, 8, 1);
        float* t = cur; cur = nxt; nxt = t;
    }

    // out = node_rep @ W_out^T + b_out   (N_OUT = 64 -> 4 tiles, 4 waves)
    k_gemm_wmma<<<mBlocks, 128, 0, stream>>>(cur, nullptr, W_out, b_out,
                                             nullptr, nullptr, out, 4, 0);
}
